// ERAFT_17781164605458
// MI455X (gfx1250) — matrix operand order repaired and checked by reading, not compiled, not run on hardware
//
#include <hip/hip_runtime.h>
#include <hip/hip_bf16.h>
#include <math.h>

// ---------------------------------------------------------------------------
// Types for CDNA5 WMMA (wave32): A/B = 16 x bf16 (8 VGPRs), C/D = 8 x f32.
// ---------------------------------------------------------------------------
typedef __attribute__((ext_vector_type(16))) __bf16          v16bf;
typedef __attribute__((ext_vector_type(16))) unsigned short  v16s;
typedef __attribute__((ext_vector_type(8)))  float           v8f;
typedef __attribute__((ext_vector_type(4)))  unsigned int    v4u;
typedef __attribute__((ext_vector_type(8)))  int             v8i;
typedef __attribute__((ext_vector_type(4)))  int             v4i;

#define THREADS 256
#define H8 44
#define W8 60
#define P8 2640          // 44*60
#define HR (352*480)     // 168960

// ---- Tensor Data Mover (TDM) availability / arity dispatch ----------------
#if __has_builtin(__builtin_amdgcn_tensor_load_to_lds)
  #define HAVE_TDM 1
  #if defined(__clang_major__) && __clang_major__ >= 23
    // amdgpu-toolchain (clang-23): 6-arg form
    #define TDM_LOAD(g0, g1, g2, g3) \
      __builtin_amdgcn_tensor_load_to_lds((g0), (g1), (g2), (g3), \
          (v8i){0,0,0,0,0,0,0,0}, 0)
  #else
    // ROCm 7.2 (clang-22): 5-arg form
    #define TDM_LOAD(g0, g1, g2, g3) \
      __builtin_amdgcn_tensor_load_to_lds((g0), (g1), (g2), (g3), 0)
  #endif
#endif

#if __has_builtin(__builtin_amdgcn_s_wait_tensorcnt)
  #define TENSOR_WAIT0() __builtin_amdgcn_s_wait_tensorcnt(0)
#else
  #define TENSOR_WAIT0() asm volatile("s_wait_tensorcnt 0x0" ::: "memory")
#endif

__device__ __forceinline__ unsigned short f2bfu(float f) {
  unsigned u = __float_as_uint(f);
  unsigned r = u + 0x7FFFu + ((u >> 16) & 1u);   // round-to-nearest-even
  return (unsigned short)(r >> 16);
}

__device__ __forceinline__ float actf(float v, int act) {
  if (act == 1) return fmaxf(v, 0.f);
  if (act == 2) return 1.f / (1.f + expf(-v));
  if (act == 3) return tanhf(v);
  return v;
}

// ---------------------------------------------------------------------------
// Implicit-GEMM convolution, bf16 WMMA, fused bias + activation + scale.
//   Out[b, m, n] = act(scale * (sum_k W[m,k] * im2col(X)[k,n] + bias[m]))
// Block tile 64(M) x 64(N), K staged 32 at a time, LDS double-buffered.
// 8 waves = 4(M) x 2(N); each wave: 16x32 C strip -> 2 WMMAs per K step.
// Weight tile staged by the Tensor Data Mover (TENSORcnt), with TDM padding
// reproducing the 40-half LDS row pitch; im2col tile staged by threads.
// ---------------------------------------------------------------------------
__global__ __launch_bounds__(THREADS) void conv_wmma(
    const unsigned short* __restrict__ Wb, const float* __restrict__ bias,
    const float* __restrict__ X, float* __restrict__ Y,
    int Cin, int IH, int IW, int OH, int OW,
    int KH, int KW, int stride, int padH, int padW,
    int M, int K, int Kp, int act, float scale)
{
  __shared__ unsigned short sA[2][64][40];   // 2 x 5120 B (row pitch 40 halves)
  __shared__ unsigned short sB[2][32][72];   // 2 x 4608 B

  const int N    = OH * OW;
  const int nb   = blockIdx.x * 64;
  const int mb   = blockIdx.y * 64;
  const int b    = blockIdx.z;
  const int tid  = threadIdx.x;
  const int lane = tid & 31;
  const int wave = tid >> 5;
  const int wm   = wave & 3;   // 16-row M sub-tile
  const int wn   = wave >> 2;  // 32-col N strip
  const int hi   = lane >> 4;
  const int KHW  = KH * KW;
  const float* Xb = X + (size_t)b * Cin * IH * IW;

  v8f acc0 = {0.f,0.f,0.f,0.f,0.f,0.f,0.f,0.f};
  v8f acc1 = {0.f,0.f,0.f,0.f,0.f,0.f,0.f,0.f};

  // ---- stage A (weights) into sA[buf] for K tile at k0 ----
  auto stageA = [&](int buf, int k0) {
#ifdef HAVE_TDM
    if (wave == 0) {
      unsigned long long ga =
          (unsigned long long)(const void*)(Wb + (size_t)mb * Kp + k0);
      unsigned lds = (unsigned)(unsigned long long)(const void*)&sA[buf][0][0];
      unsigned td0 = (unsigned)(Kp - k0);                  // elements left in K
      unsigned td1 = (M > mb) ? (unsigned)(M - mb) : 0u;   // rows left (OOB->0)
      v4u g0 = { 1u,                                       // count=1 (valid D#)
                 lds,                                      // lds_addr
                 (unsigned)ga,                             // global_addr[31:0]
                 (unsigned)((ga >> 32) & 0x01FFFFFFu) | (2u << 30) }; // +type=2
      v8i g1 = { (int)((1u << 16)      // data_size = 2B
                     | (1u << 20)      // pad_enable
                     | (3u << 22)      // pad_interval: every 16 DWORDs
                     | (3u << 25)),    // pad_amount : 4 DWORDs (8 halves)
                 (int)((td0 & 0xFFFFu) << 16),             // tensor_dim0 lo16
                 (int)((td0 >> 16) | ((td1 & 0xFFFFu) << 16)),
                 (int)((td1 >> 16) | (32u << 16)),         // tile_dim0 = 32
                 (int)64,                                  // tile_dim1 = 64
                 (int)(unsigned)Kp,                        // dim0 stride (elems)
                 0, 0 };
      v4i gz = { 0, 0, 0, 0 };
      TDM_LOAD(g0, g1, gz, gz);
    }
#else
    #pragma unroll
    for (int e = tid; e < 64 * 32; e += THREADS) {
      int m = e >> 5, k = e & 31;
      int gm = mb + m;
      sA[buf][m][k] = (gm < M) ? Wb[(size_t)gm * Kp + (k0 + k)]
                               : (unsigned short)0;
    }
#endif
  };

  // ---- stage B (im2col activations) 32(K) x 64(N) into sB[buf] ----
  auto stageB = [&](int buf, int k0) {
    #pragma unroll
    for (int e = tid; e < 32 * 64; e += THREADS) {
      int k = e >> 6, n = e & 63;
      int gn = nb + n, gk = k0 + k;
      float v = 0.f;
      if (gn < N && gk < K) {
        int ci  = gk / KHW;
        int rem = gk - ci * KHW;
        int ky  = rem / KW;
        int kx  = rem - ky * KW;
        int oy  = gn / OW;
        int ox  = gn - oy * OW;
        int iy  = oy * stride - padH + ky;
        int ix  = ox * stride - padW + kx;
        if (iy >= 0 && iy < IH && ix >= 0 && ix < IW)
          v = Xb[((size_t)ci * IH + iy) * IW + ix];
      }
      sB[buf][k][n] = f2bfu(v);
    }
  };

  const int NK = Kp >> 5;

  stageA(0, 0);
  stageB(0, 0);
#ifdef HAVE_TDM
  if (wave == 0) TENSOR_WAIT0();
#endif
  __syncthreads();

  for (int kt = 0; kt < NK; ++kt) {
    const int buf = kt & 1;
    if (kt + 1 < NK) {            // overlap next-tile staging with compute
      stageA(buf ^ 1, (kt + 1) << 5);
      stageB(buf ^ 1, (kt + 1) << 5);
    }

    // ---- fragments: contiguous k = 16*hi + i in both A and B ----
    v16s av, bv0, bv1;
    const int mrow = wm * 16 + (lane & 15);
    #pragma unroll
    for (int i = 0; i < 16; ++i) {
      int ka = 16 * hi + i;
      av[i] = sA[buf][mrow][ka];
    }
    const int ncol = wn * 32 + (lane & 15);
    const int kb0  = hi * 16;
    #pragma unroll
    for (int i = 0; i < 16; ++i) {
      bv0[i] = sB[buf][kb0 + i][ncol];
      bv1[i] = sB[buf][kb0 + i][ncol + 16];
    }

    acc0 = __builtin_amdgcn_wmma_f32_16x16x32_bf16(
        false, __builtin_bit_cast(v16bf, av),
        false, __builtin_bit_cast(v16bf, bv0),
        (short)0, acc0, false, false);
    acc1 = __builtin_amdgcn_wmma_f32_16x16x32_bf16(
        false, __builtin_bit_cast(v16bf, av),
        false, __builtin_bit_cast(v16bf, bv1),
        (short)0, acc1, false, false);

#ifdef HAVE_TDM
    if (wave == 0) TENSOR_WAIT0();
#endif
    __syncthreads();
  }

  // ---- epilogue: bias + scale + activation ----
  const int n0 = nb + wn * 32 + (lane & 15);
  const int n1 = n0 + 16;
  #pragma unroll
  for (int r = 0; r < 8; ++r) {
    int m = mb + wm * 16 + r + 8 * hi;
    if (m < M) {
      float bsv = bias ? bias[m] : 0.f;
      if (n0 < N)
        Y[((size_t)b * M + m) * N + n0] = actf((acc0[r] + bsv) * scale, act);
      if (n1 < N)
        Y[((size_t)b * M + m) * N + n1] = actf((acc1[r] + bsv) * scale, act);
    }
  }
}

// ---------------------------------------------------------------------------
// Correlation volume: corr[p,q] = (1/16) * sum_d f1[d,p] * f2[d,q]
// Block tile 64(M) x 64(N), K = D = 256. A is f1 transposed on the fly.
// ---------------------------------------------------------------------------
__global__ __launch_bounds__(THREADS) void corr_wmma(
    const float* __restrict__ f1, const float* __restrict__ f2,
    float* __restrict__ corr, int P, int D)
{
  __shared__ unsigned short sA[64][40];
  __shared__ unsigned short sB[32][72];

  const int nb   = blockIdx.x * 64;
  const int mb   = blockIdx.y * 64;
  const int tid  = threadIdx.x;
  const int lane = tid & 31;
  const int wave = tid >> 5;
  const int wm   = wave & 3;
  const int wn   = wave >> 2;
  const int hi   = lane >> 4;

  v8f acc0 = {0.f,0.f,0.f,0.f,0.f,0.f,0.f,0.f};
  v8f acc1 = {0.f,0.f,0.f,0.f,0.f,0.f,0.f,0.f};

  for (int k0 = 0; k0 < D; k0 += 32) {
    #pragma unroll
    for (int e = tid; e < 64 * 32; e += THREADS) {
      int m = e >> 5, k = e & 31;
      int gm = mb + m;
      sA[m][k] = (gm < P) ? f2bfu(f1[(size_t)(k0 + k) * P + gm])
                          : (unsigned short)0;
    }
    #pragma unroll
    for (int e = tid; e < 32 * 64; e += THREADS) {
      int k = e >> 6, n = e & 63;
      int gn = nb + n;
      sB[k][n] = (gn < P) ? f2bfu(f2[(size_t)(k0 + k) * P + gn])
                          : (unsigned short)0;
    }
    __syncthreads();

    v16s av, bv0, bv1;
    const int mrow = wm * 16 + (lane & 15);
    #pragma unroll
    for (int i = 0; i < 16; ++i) {
      int ka = 16 * hi + i;
      av[i] = sA[mrow][ka];
    }
    const int ncol = wn * 32 + (lane & 15);
    const int kb0  = hi * 16;
    #pragma unroll
    for (int i = 0; i < 16; ++i) {
      bv0[i] = sB[kb0 + i][ncol];
      bv1[i] = sB[kb0 + i][ncol + 16];
    }

    acc0 = __builtin_amdgcn_wmma_f32_16x16x32_bf16(
        false, __builtin_bit_cast(v16bf, av),
        false, __builtin_bit_cast(v16bf, bv0),
        (short)0, acc0, false, false);
    acc1 = __builtin_amdgcn_wmma_f32_16x16x32_bf16(
        false, __builtin_bit_cast(v16bf, av),
        false, __builtin_bit_cast(v16bf, bv1),
        (short)0, acc1, false, false);
    __syncthreads();
  }

  const int n0 = nb + wn * 32 + (lane & 15);
  const int n1 = n0 + 16;
  #pragma unroll
  for (int r = 0; r < 8; ++r) {
    int m = mb + wm * 16 + r + 8 * hi;
    if (m < P) {
      if (n0 < P) corr[(size_t)m * P + n0] = acc0[r] * 0.0625f;
      if (n1 < P) corr[(size_t)m * P + n1] = acc1[r] * 0.0625f;
    }
  }
}

// ---------------------------------------------------------------------------
// Pack fp32 OIHW weights -> bf16 [M][Kp] with zero K-padding.
// ---------------------------------------------------------------------------
__global__ void pack_w(const float* __restrict__ w, unsigned short* __restrict__ o,
                       int M, int K, int Kp)
{
  int i = blockIdx.x * THREADS + threadIdx.x;
  if (i >= M * Kp) return;
  int m = i / Kp, k = i - m * Kp;
  o[i] = f2bfu(k < K ? w[(size_t)m * K + k] : 0.f);
}

// ---------------------------------------------------------------------------
// Instance/Batch norm (B==1 batch norm == per (n,c) reduction) + opt affine/relu
// ---------------------------------------------------------------------------
__global__ __launch_bounds__(THREADS) void k_norm(
    float* __restrict__ x, const float* __restrict__ g, const float* __restrict__ bta,
    int C, int HW, int do_relu)
{
  __shared__ float ss[THREADS], ss2[THREADS];
  int nc = blockIdx.x;
  int c  = nc % C;
  float* p = x + (size_t)nc * HW;
  float s = 0.f, s2 = 0.f;
  for (int i = threadIdx.x; i < HW; i += THREADS) { float v = p[i]; s += v; s2 += v * v; }
  ss[threadIdx.x] = s; ss2[threadIdx.x] = s2;
  __syncthreads();
  for (int o = THREADS / 2; o > 0; o >>= 1) {
    if (threadIdx.x < o) { ss[threadIdx.x] += ss[threadIdx.x + o]; ss2[threadIdx.x] += ss2[threadIdx.x + o]; }
    __syncthreads();
  }
  float mean = ss[0] / HW;
  float var  = ss2[0] / HW - mean * mean;
  float rs   = rsqrtf(var + 1e-5f);
  float ga = g ? g[c] : 1.f, be = bta ? bta[c] : 0.f;
  for (int i = threadIdx.x; i < HW; i += THREADS) {
    float v = (p[i] - mean) * rs * ga + be;
    if (do_relu) v = fmaxf(v, 0.f);
    p[i] = v;
  }
}

// ---------------------------------------------------------------------------
// Small elementwise kernels
// ---------------------------------------------------------------------------
__global__ void k_copy(float* __restrict__ d, const float* __restrict__ s, int n)
{ int i = blockIdx.x * THREADS + threadIdx.x; if (i < n) d[i] = s[i]; }

__global__ void k_zero(float* __restrict__ d, int n)
{ int i = blockIdx.x * THREADS + threadIdx.x; if (i < n) d[i] = 0.f; }

__global__ void k_add_relu(float* __restrict__ d, const float* __restrict__ a,
                           const float* __restrict__ b, int n)
{ int i = blockIdx.x * THREADS + threadIdx.x; if (i < n) d[i] = fmaxf(a[i] + b[i], 0.f); }

__global__ void k_addto(float* __restrict__ d, const float* __restrict__ a, int n)
{ int i = blockIdx.x * THREADS + threadIdx.x; if (i < n) d[i] += a[i]; }

__global__ void k_mul(float* __restrict__ d, const float* __restrict__ a,
                      const float* __restrict__ b, int n)
{ int i = blockIdx.x * THREADS + threadIdx.x; if (i < n) d[i] = a[i] * b[i]; }

__global__ void k_gru(float* __restrict__ h, const float* __restrict__ z,
                      const float* __restrict__ q, int n)
{ int i = blockIdx.x * THREADS + threadIdx.x; if (i < n) h[i] = (1.f - z[i]) * h[i] + z[i] * q[i]; }

__global__ void k_split_cnet(const float* __restrict__ c, float* __restrict__ net,
                             float* __restrict__ inp)
{
  int i = blockIdx.x * THREADS + threadIdx.x;
  if (i >= 128 * P8) return;
  net[i] = tanhf(c[i]);
  float x = c[128 * P8 + i];
  inp[i] = 0.5f * x * (1.f + erff(x * 0.70710678f));   // exact GELU
}

__global__ void k_pool(const float* __restrict__ in, float* __restrict__ out,
                       int P, int ih, int iw, int oh, int ow)
{
  int i = blockIdx.x * THREADS + threadIdx.x;
  if (i >= P * oh * ow) return;
  int p = i / (oh * ow), r = i - p * (oh * ow);
  int y = r / ow, x = r - y * ow;
  const float* s = in + (size_t)p * ih * iw + (2 * y) * iw + 2 * x;
  out[i] = 0.25f * (s[0] + s[1] + s[iw] + s[iw + 1]);
}

// Correlation lookup: out[(lev*81+j), p] = bilinear(pyr[lev][p], c/2^lev + d)
__global__ void k_lookup(const float* __restrict__ flow,
                         const float* __restrict__ c0, const float* __restrict__ c1,
                         const float* __restrict__ c2, const float* __restrict__ c3,
                         float* __restrict__ out)
{
  int idx = blockIdx.x * THREADS + threadIdx.x;
  if (idx >= 324 * P8) return;
  int ch = idx / P8, p = idx - ch * P8;
  int py = p / W8, px = p - py * W8;
  float cx = (float)px + flow[p];
  float cy = (float)py + flow[P8 + p];
  int lev = ch / 81, j = ch - lev * 81;
  float dxo = (float)(j % 9) - 4.f;
  float dyo = (float)(j / 9) - 4.f;
  const float* base; int hl, wl;
  if (lev == 0)      { base = c0; hl = 44; wl = 60; }
  else if (lev == 1) { base = c1; hl = 22; wl = 30; }
  else if (lev == 2) { base = c2; hl = 11; wl = 15; }
  else               { base = c3; hl = 5;  wl = 7;  }
  float sc = 1.f / (float)(1 << lev);
  float xs = cx * sc + dxo, ys = cy * sc + dyo;
  const float* img = base + (size_t)p * hl * wl;
  float x0f = floorf(xs), y0f = floorf(ys);
  float wx = xs - x0f, wy = ys - y0f;
  int x0 = (int)x0f, y0 = (int)y0f;
  auto g = [&](int xi, int yi) -> float {
    if (xi < 0 || xi > wl - 1 || yi < 0 || yi > hl - 1) return 0.f;
    return img[yi * wl + xi];
  };
  out[idx] = g(x0, y0) * (1.f - wx) * (1.f - wy) + g(x0 + 1, y0) * wx * (1.f - wy)
           + g(x0, y0 + 1) * (1.f - wx) * wy     + g(x0 + 1, y0 + 1) * wx * wy;
}

// Convex upsampling (8x): softmax(mask) over 3x3 neighborhood of 8*flow
__global__ void k_upsample(const float* __restrict__ flow, const float* __restrict__ mask,
                           float* __restrict__ ou, float* __restrict__ ov)
{
  int idx = blockIdx.x * THREADS + threadIdx.x;
  if (idx >= HR) return;
  int Y = idx / 480, X = idx - Y * 480;
  int h = Y >> 3, r = Y & 7, w = X >> 3, s = X & 7;
  int p = h * W8 + w;
  float e[9]; float mx = -3.4e38f;
  #pragma unroll
  for (int j = 0; j < 9; ++j) {
    float m = mask[(size_t)(j * 64 + r * 8 + s) * P8 + p];
    e[j] = m; mx = fmaxf(mx, m);
  }
  float sum = 0.f;
  #pragma unroll
  for (int j = 0; j < 9; ++j) { e[j] = expf(e[j] - mx); sum += e[j]; }
  float inv = 1.f / sum;
  float fu = 0.f, fv = 0.f;
  #pragma unroll
  for (int j = 0; j < 9; ++j) {
    int ky = j / 3 - 1, kx = j % 3 - 1;
    int hy = h + ky, wx2 = w + kx;
    float pu = 0.f, pv = 0.f;
    if (hy >= 0 && hy < H8 && wx2 >= 0 && wx2 < W8) {
      pu = 8.f * flow[hy * W8 + wx2];
      pv = 8.f * flow[P8 + hy * W8 + wx2];
    }
    float pj = e[j] * inv;
    fu += pj * pu; fv += pj * pv;
  }
  ou[idx] = fu; ov[idx] = fv;
}

// ===========================================================================
// Host orchestration
// ===========================================================================
namespace {

struct Bump {
  char* base; size_t off, cap;
  void* take(size_t bytes) {
    size_t a = (off + 255) & ~(size_t)255;
    off = a + bytes;
    return base + a;
  }
  float* f(size_t n) { return (float*)take(n * sizeof(float)); }
  unsigned short* h(size_t n) { return (unsigned short*)take(n * sizeof(unsigned short)); }
};

struct ConvW { const unsigned short* w; int M, K, Kp; };

static inline int ceil_div(int a, int b) { return (a + b - 1) / b; }

static ConvW pack(hipStream_t st, Bump& ws, const float* w, int M, int K) {
  int Kp = (K + 31) & ~31;
  unsigned short* o = ws.h((size_t)M * Kp);
  int n = M * Kp;
  pack_w<<<ceil_div(n, THREADS), THREADS, 0, st>>>(w, o, M, K, Kp);
  ConvW cw; cw.w = o; cw.M = M; cw.K = K; cw.Kp = Kp;
  return cw;
}

static void conv(hipStream_t st, const ConvW& cw, const float* bias,
                 const float* X, float* Y, int B, int Cin,
                 int IH, int IW, int OH, int OW, int KH, int KW,
                 int s, int pH, int pW, int act, float scale = 1.f) {
  dim3 g(ceil_div(OH * OW, 64), ceil_div(cw.M, 64), B);
  conv_wmma<<<g, THREADS, 0, st>>>(cw.w, bias, X, Y, Cin, IH, IW, OH, OW,
                                   KH, KW, s, pH, pW, cw.M, cw.K, cw.Kp, act, scale);
}

static void norm(hipStream_t st, float* x, const float* g, const float* b,
                 int Bn, int C, int HW, int relu) {
  k_norm<<<Bn * C, THREADS, 0, st>>>(x, g, b, C, HW, relu);
}

static inline void cpy(hipStream_t st, float* d, const float* s, int n) {
  k_copy<<<ceil_div(n, THREADS), THREADS, 0, st>>>(d, s, n);
}

// Param leaf indices (JAX pytree flattening: dict keys sorted alphabetically).
struct BlkIdx { int b1, b2, c1, c2, bd, cd, n1g, n1b, n2g, n2b, ndg, ndb; };
struct EncIdx { int conv1w, conv1b, n1g, n1b, conv2w, conv2b; BlkIdx blk[6]; };

static const EncIdx CNET = {
  59, 58, 63, 62, 61, 60,
  { {  2,  3,  4,  5, -1, -1,  7,  6,  9,  8, -1, -1 },
    { 10, 11, 12, 13, -1, -1, 15, 14, 17, 16, -1, -1 },
    { 18, 19, 21, 22, 20, 23, 25, 24, 27, 26, 29, 28 },
    { 30, 31, 32, 33, -1, -1, 35, 34, 37, 36, -1, -1 },
    { 38, 39, 41, 42, 40, 43, 45, 44, 47, 46, 49, 48 },
    { 50, 51, 52, 53, -1, -1, 55, 54, 57, 56, -1, -1 } }
};
static const EncIdx FNET = {
  93, 92, -1, -1, 95, 94,
  { { 64, 65, 66, 67, -1, -1, -1, -1, -1, -1, -1, -1 },
    { 68, 69, 70, 71, -1, -1, -1, -1, -1, -1, -1, -1 },
    { 72, 73, 75, 76, 74, 77, -1, -1, -1, -1, -1, -1 },
    { 78, 79, 80, 81, -1, -1, -1, -1, -1, -1, -1, -1 },
    { 82, 83, 85, 86, 84, 87, -1, -1, -1, -1, -1, -1 },
    { 88, 89, 90, 91, -1, -1, -1, -1, -1, -1, -1, -1 } }
};

static void run_encoder(hipStream_t st, Bump& ws, const float* const* in,
                        const EncIdx& E, int B, const float* x0,
                        float* bufs[4], float* out) {
  const float* g1 = E.n1g >= 0 ? in[E.n1g] : nullptr;
  const float* b1 = E.n1b >= 0 ? in[E.n1b] : nullptr;

  ConvW w1 = pack(st, ws, in[E.conv1w], 64, 5 * 49);
  conv(st, w1, in[E.conv1b], x0, bufs[0], B, 5, 352, 480, 176, 240, 7, 7, 2, 3, 3, 0);
  norm(st, bufs[0], g1, b1, B, 64, 176 * 240, 1);

  float* cur = bufs[0];
  int H = 176, W = 240;
  const int ci[6] = { 64, 64, 64, 96, 96, 128 };
  const int co[6] = { 64, 64, 96, 96, 128, 128 };
  const int ss[6] = { 1, 1, 2, 1, 2, 1 };

  for (int i = 0; i < 6; ++i) {
    const BlkIdx& Bk = E.blk[i];
    float* rem[3]; int rr = 0;
    for (int j = 0; j < 4; ++j) if (bufs[j] != cur) rem[rr++] = bufs[j];
    float* t1 = rem[0]; float* t2 = rem[1]; float* t3 = rem[2];

    int s = ss[i];
    int OH = (H + 2 - 3) / s + 1, OW = (W + 2 - 3) / s + 1;

    ConvW c1 = pack(st, ws, in[Bk.c1], co[i], ci[i] * 9);
    conv(st, c1, in[Bk.b1], cur, t1, B, ci[i], H, W, OH, OW, 3, 3, s, 1, 1, 0);
    norm(st, t1, Bk.n1g >= 0 ? in[Bk.n1g] : nullptr,
         Bk.n1b >= 0 ? in[Bk.n1b] : nullptr, B, co[i], OH * OW, 1);

    ConvW c2 = pack(st, ws, in[Bk.c2], co[i], co[i] * 9);
    conv(st, c2, in[Bk.b2], t1, t2, B, co[i], OH, OW, OH, OW, 3, 3, 1, 1, 1, 0);
    norm(st, t2, Bk.n2g >= 0 ? in[Bk.n2g] : nullptr,
         Bk.n2b >= 0 ? in[Bk.n2b] : nullptr, B, co[i], OH * OW, 1);

    const float* px = cur;
    if (Bk.cd >= 0) {
      ConvW cd = pack(st, ws, in[Bk.cd], co[i], ci[i]);
      conv(st, cd, in[Bk.bd], cur, t3, B, ci[i], H, W, OH, OW, 1, 1, s, 0, 0, 0);
      norm(st, t3, Bk.ndg >= 0 ? in[Bk.ndg] : nullptr,
           Bk.ndb >= 0 ? in[Bk.ndb] : nullptr, B, co[i], OH * OW, 0);
      px = t3;
    }
    int n = B * co[i] * OH * OW;
    k_add_relu<<<ceil_div(n, THREADS), THREADS, 0, st>>>(t1, px, t2, n);
    cur = t1; H = OH; W = OW;
  }
  ConvW w2 = pack(st, ws, in[E.conv2w], 256, 128);
  conv(st, w2, in[E.conv2b], cur, out, B, 128, H, W, H, W, 1, 1, 1, 0, 0, 0);
}

} // namespace

extern "C" void kernel_launch(void* const* d_in, const int* in_sizes, int n_in,
                              void* d_out, int out_size, void* d_ws, size_t ws_size,
                              hipStream_t stream) {
  (void)in_sizes; (void)n_in; (void)out_size;
  const float* const* in = (const float* const*)d_in;
  float* out = (float*)d_out;
  hipStream_t st = stream;

  Bump ws; ws.base = (char*)d_ws; ws.off = 0; ws.cap = ws_size;

  // ---------------- buffers ----------------
  const size_t ENC = (size_t)2 * 64 * 176 * 240;
  float* xcat   = ws.f((size_t)2 * 5 * 352 * 480);
  float* bufs[4] = { ws.f(ENC), ws.f(ENC), ws.f(ENC), ws.f(ENC) };
  float* fmap   = ws.f((size_t)2 * 256 * P8);
  float* cnetO  = ws.f((size_t)256 * P8);
  float* pyr0   = ws.f((size_t)P8 * 44 * 60);
  float* pyr1   = ws.f((size_t)P8 * 22 * 30);
  float* pyr2   = ws.f((size_t)P8 * 11 * 15);
  float* pyr3   = ws.f((size_t)P8 * 5 * 7);
  float* net    = ws.f((size_t)128 * P8);
  float* inp    = ws.f((size_t)128 * P8);
  float* flow   = ws.f((size_t)2 * P8);
  float* dflow  = ws.f((size_t)2 * P8);
  float* corrf  = ws.f((size_t)324 * P8);
  float* cor    = ws.f((size_t)256 * P8);
  float* cor2   = ws.f((size_t)192 * P8);
  float* flo    = ws.f((size_t)128 * P8);
  float* flo2   = ws.f((size_t)64 * P8);
  float* cf     = ws.f((size_t)256 * P8);
  float* mf     = ws.f((size_t)126 * P8);
  float* xbuf   = ws.f((size_t)256 * P8);
  float* hx     = ws.f((size_t)384 * P8);
  float* zb     = ws.f((size_t)128 * P8);
  float* rb     = ws.f((size_t)128 * P8);
  float* qin    = ws.f((size_t)384 * P8);
  float* qb     = ws.f((size_t)128 * P8);
  float* tb     = ws.f((size_t)256 * P8);
  float* maskb  = ws.f((size_t)576 * P8);

  // ---------------- feature encoder (instance norm) on [event1; event2] ------
  const int IMG = 5 * 352 * 480;
  cpy(st, xcat, in[0], IMG);
  cpy(st, xcat + IMG, in[1], IMG);
  run_encoder(st, ws, in, FNET, 2, xcat, bufs, fmap);

  // ---------------- correlation pyramid ----------------
  const float* f1 = fmap;
  const float* f2 = fmap + (size_t)256 * P8;
  {
    dim3 g(ceil_div(P8, 64), ceil_div(P8, 64), 1);
    corr_wmma<<<g, THREADS, 0, st>>>(f1, f2, pyr0, P8, 256);
  }
  k_pool<<<ceil_div(P8 * 22 * 30, THREADS), THREADS, 0, st>>>(pyr0, pyr1, P8, 44, 60, 22, 30);
  k_pool<<<ceil_div(P8 * 11 * 15, THREADS), THREADS, 0, st>>>(pyr1, pyr2, P8, 22, 30, 11, 15);
  k_pool<<<ceil_div(P8 * 5 * 7,  THREADS), THREADS, 0, st>>>(pyr2, pyr3, P8, 11, 15, 5, 7);

  // ---------------- context encoder (batch norm) on event2 ----------------
  run_encoder(st, ws, in, CNET, 1, in[1], bufs, cnetO);
  k_split_cnet<<<ceil_div(128 * P8, THREADS), THREADS, 0, st>>>(cnetO, net, inp);

  k_zero<<<ceil_div(2 * P8, THREADS), THREADS, 0, st>>>(flow, 2 * P8);

  // ---------------- pack update-block weights once ----------------
  ConvW Wcc1 = pack(st, ws, in[97], 256, 324);
  ConvW Wcc2 = pack(st, ws, in[99], 192, 256 * 9);
  ConvW Wcf1 = pack(st, ws, in[101], 128, 2 * 49);
  ConvW Wcf2 = pack(st, ws, in[103], 64, 128 * 9);
  ConvW Wfh1 = pack(st, ws, in[105], 256, 128 * 9);
  ConvW Wfh2 = pack(st, ws, in[107], 2, 256 * 9);
  ConvW Wm1  = pack(st, ws, in[109], 256, 128 * 9);
  ConvW Wm2  = pack(st, ws, in[111], 576, 256);
  ConvW Wmc  = pack(st, ws, in[113], 126, 256 * 9);
  ConvW Wq1  = pack(st, ws, in[115], 128, 384 * 5);
  ConvW Wq2  = pack(st, ws, in[117], 128, 384 * 5);
  ConvW Wr1  = pack(st, ws, in[119], 128, 384 * 5);
  ConvW Wr2  = pack(st, ws, in[121], 128, 384 * 5);
  ConvW Wz1  = pack(st, ws, in[123], 128, 384 * 5);
  ConvW Wz2  = pack(st, ws, in[125], 128, 384 * 5);

  const int EW = ceil_div(128 * P8, THREADS);

  // ---------------- 12 GRU update iterations ----------------
  for (int it = 0; it < 12; ++it) {
    k_lookup<<<ceil_div(324 * P8, THREADS), THREADS, 0, st>>>(flow, pyr0, pyr1, pyr2, pyr3, corrf);

    conv(st, Wcc1, in[96],  corrf, cor,  1, 324, H8, W8, H8, W8, 1, 1, 1, 0, 0, 1);
    conv(st, Wcc2, in[98],  cor,   cor2, 1, 256, H8, W8, H8, W8, 3, 3, 1, 1, 1, 1);
    conv(st, Wcf1, in[100], flow,  flo,  1, 2,   H8, W8, H8, W8, 7, 7, 1, 3, 3, 1);
    conv(st, Wcf2, in[102], flo,   flo2, 1, 128, H8, W8, H8, W8, 3, 3, 1, 1, 1, 1);

    cpy(st, cf, cor2, 192 * P8);
    cpy(st, cf + 192 * P8, flo2, 64 * P8);
    conv(st, Wmc, in[112], cf, mf, 1, 256, H8, W8, H8, W8, 3, 3, 1, 1, 1, 1);

    // x = [inp(128), mf(126), flow(2)]
    cpy(st, xbuf, inp, 128 * P8);
    cpy(st, xbuf + 128 * P8, mf, 126 * P8);
    cpy(st, xbuf + 254 * P8, flow, 2 * P8);

    // -------- GRU horizontal (1x5) --------
    cpy(st, hx, net, 128 * P8);
    cpy(st, hx + 128 * P8, xbuf, 256 * P8);
    conv(st, Wz1, in[122], hx, zb, 1, 384, H8, W8, H8, W8, 1, 5, 1, 0, 2, 2);
    conv(st, Wr1, in[118], hx, rb, 1, 384, H8, W8, H8, W8, 1, 5, 1, 0, 2, 2);
    k_mul<<<EW, THREADS, 0, st>>>(qin, rb, net, 128 * P8);
    cpy(st, qin + 128 * P8, xbuf, 256 * P8);
    conv(st, Wq1, in[114], qin, qb, 1, 384, H8, W8, H8, W8, 1, 5, 1, 0, 2, 3);
    k_gru<<<EW, THREADS, 0, st>>>(net, zb, qb, 128 * P8);

    // -------- GRU vertical (5x1) --------
    cpy(st, hx, net, 128 * P8);
    cpy(st, hx + 128 * P8, xbuf, 256 * P8);
    conv(st, Wz2, in[124], hx, zb, 1, 384, H8, W8, H8, W8, 5, 1, 1, 2, 0, 2);
    conv(st, Wr2, in[120], hx, rb, 1, 384, H8, W8, H8, W8, 5, 1, 1, 2, 0, 2);
    k_mul<<<EW, THREADS, 0, st>>>(qin, rb, net, 128 * P8);
    cpy(st, qin + 128 * P8, xbuf, 256 * P8);
    conv(st, Wq2, in[116], qin, qb, 1, 384, H8, W8, H8, W8, 5, 1, 1, 2, 0, 3);
    k_gru<<<EW, THREADS, 0, st>>>(net, zb, qb, 128 * P8);

    // -------- flow head + mask head --------
    conv(st, Wfh1, in[104], net, tb, 1, 128, H8, W8, H8, W8, 3, 3, 1, 1, 1, 1);
    conv(st, Wfh2, in[106], tb, dflow, 1, 256, H8, W8, H8, W8, 3, 3, 1, 1, 1, 0);
    conv(st, Wm1, in[108], net, tb, 1, 128, H8, W8, H8, W8, 3, 3, 1, 1, 1, 1);
    conv(st, Wm2, in[110], tb, maskb, 1, 256, H8, W8, H8, W8, 1, 1, 1, 0, 0, 0, 0.25f);

    k_addto<<<ceil_div(2 * P8, THREADS), THREADS, 0, st>>>(flow, dflow, 2 * P8);

    float* ou = out + 2 * P8 + (size_t)it * (2 * HR);
    k_upsample<<<ceil_div(HR, THREADS), THREADS, 0, st>>>(flow, maskb, ou, ou + HR);
  }

  // output[0] = low-res flow (1,2,44,60)
  cpy(st, out, flow, 2 * P8);
}
